// LSTM_14242111554073
// MI455X (gfx1250) — matrix operand order repaired and checked by reading, not compiled, not run on hardware
//
#include <hip/hip_runtime.h>

typedef __attribute__((ext_vector_type(16))) __bf16 v16bf;
typedef __attribute__((ext_vector_type(8)))  __bf16 v8bf;
typedef __attribute__((ext_vector_type(8)))  float  v8f;

#define B_   32
#define S_   512
#define IN_  512
#define H_   1024
#define NB   16     // blocks in the persistent scan kernel (must all be co-resident)

// ---- workspace layout (bytes) ----
// gx fragments  : 512 steps * 2 Mtiles * 256 Ntiles * 32 lanes * 8 bf16 = 128 MB
#define OFF_GX   0UL
#define OFF_WXT  (OFF_GX  + 512UL*2*256*32*8*2)   // wxT bf16 [4096][512]   : 4 MB
#define OFF_WHT  (OFF_WXT + 4096UL*512*2)         // whT bf16 [4096][1024]  : 8 MB
#define OFF_XB   (OFF_WHT + 4096UL*1024*2)        // x bf16  [32][512][512] : 16 MB
#define OFF_BH   (OFF_XB  + 32UL*512*512*2)       // bias f32 [4096]        : 16 KB
#define OFF_H    (OFF_BH  + 4096UL*4)             // h bf16, double-buffered: 128 KB
#define OFF_CNT  (OFF_H   + 2UL*32*1024*2)        // barrier counter

__device__ __forceinline__ float fast_sigmoid(float x) {
    return 1.0f / (1.0f + __expf(-x));
}

// ---------------- packing / init kernels ----------------
__global__ void pack_wx(const float* __restrict__ f, const float* __restrict__ i,
                        const float* __restrict__ o, const float* __restrict__ c,
                        __bf16* __restrict__ wxT) {
    size_t idx = (size_t)blockIdx.x * blockDim.x + threadIdx.x;   // g*512 + k
    if (idx >= 4096UL * 512UL) return;
    int g = (int)(idx >> 9), k = (int)(idx & 511);
    const float* src = (g < 1024) ? f : (g < 2048) ? i : (g < 3072) ? o : c;
    wxT[idx] = (__bf16)src[(size_t)k * 1024 + (g & 1023)];
}

__global__ void pack_wh(const float* __restrict__ f, const float* __restrict__ i,
                        const float* __restrict__ o, const float* __restrict__ c,
                        __bf16* __restrict__ whT) {
    size_t idx = (size_t)blockIdx.x * blockDim.x + threadIdx.x;   // g*1024 + k
    if (idx >= 4096UL * 1024UL) return;
    int g = (int)(idx >> 10), k = (int)(idx & 1023);
    const float* src = (g < 1024) ? f : (g < 2048) ? i : (g < 3072) ? o : c;
    whT[idx] = (__bf16)src[(size_t)k * 1024 + (g & 1023)];
}

__global__ void pack_x(const float* __restrict__ x, __bf16* __restrict__ xb) {
    size_t idx = (size_t)blockIdx.x * blockDim.x + threadIdx.x;
    if (idx >= 32UL * 512UL * 512UL) return;
    xb[idx] = (__bf16)x[idx];
}

__global__ void pack_misc(const float* __restrict__ bf, const float* __restrict__ bi,
                          const float* __restrict__ bo, const float* __restrict__ bc,
                          float* __restrict__ bh4, __bf16* __restrict__ hbuf,
                          unsigned* __restrict__ cnt) {
    int idx = blockIdx.x * blockDim.x + threadIdx.x;
    if (idx < 4096) {
        const float* src = (idx < 1024) ? bf : (idx < 2048) ? bi : (idx < 3072) ? bo : bc;
        bh4[idx] = src[idx & 1023];
    }
    if (idx < 2 * 32 * 1024) hbuf[idx] = (__bf16)0.0f;   // zero both h buffers
    if (idx == 0) *cnt = 0u;                              // reset grid barrier
}

// ---------------- phase 1: gx = x @ wx + bias (bf16 WMMA, f32 acc) ----------------
// one 16x16 C tile per wave; C stored in fragment layout (per-lane 8 bf16 = 16B)
__global__ void gx_gemm(const __bf16* __restrict__ xb, const __bf16* __restrict__ wxT,
                        const float* __restrict__ bh4, __bf16* __restrict__ gxf) {
    int wave = (blockIdx.x * 256 + threadIdx.x) >> 5;
    int lane = threadIdx.x & 31;
    int n = wave & 255;          // N tile over 4096 cols
    int m = (wave >> 8) & 1;     // M tile over 32 batch rows
    int s = wave >> 9;           // sequence index
    if (s >= S_) return;
    int hi = lane >> 4, lm = lane & 15;
    int g = n * 16 + lm;

    v8f acc;
    float bias = bh4[g];
    #pragma unroll
    for (int j = 0; j < 8; ++j) acc[j] = bias;

    int brow = m * 16 + lm;
    const __bf16* arow = xb  + ((size_t)brow * S_ + s) * IN_;
    const __bf16* bcol = wxT + (size_t)g * IN_;

    #pragma unroll 4
    for (int kk = 0; kk < IN_ / 32; ++kk) {
        int k0 = kk * 32 + hi * 8;
        v8bf a0 = *(const v8bf*)(arow + k0);
        v8bf a1 = *(const v8bf*)(arow + k0 + 16);
        v16bf a;
        #pragma unroll
        for (int j = 0; j < 8; ++j) { a[j] = a0[j]; a[j + 8] = a1[j]; }
        int kb = kk * 32 + hi * 8;
        v8bf b0 = *(const v8bf*)(bcol + kb);
        v8bf b1 = *(const v8bf*)(bcol + kb + 16);
        v16bf b;
        #pragma unroll
        for (int j = 0; j < 8; ++j) { b[j] = b0[j]; b[j + 8] = b1[j]; }
        acc = __builtin_amdgcn_wmma_f32_16x16x32_bf16(false, a, false, b,
                                                      (short)0, acc, false, false);
    }
    size_t idx = ((((size_t)s * 2 + m) * 256 + n) * 32 + lane) * 8;
    v8bf st;
    #pragma unroll
    for (int j = 0; j < 8; ++j) st[j] = (__bf16)acc[j];
    *(v8bf*)(gxf + idx) = st;
}

// ---------------- phase 2: persistent recurrent scan ----------------
// 128 waves; each wave owns all 4 gate tiles for a (16 batch x 16 hcol) slice.
// c stays in registers across all 512 steps; h double-buffered in L2.
__global__ void lstm_steps(const __bf16* __restrict__ gxf, const __bf16* __restrict__ whT,
                           __bf16* __restrict__ hbuf, float* __restrict__ out,
                           unsigned* __restrict__ cnt) {
    int wave = (blockIdx.x * 256 + threadIdx.x) >> 5;   // 0..127
    int lane = threadIdx.x & 31;
    int m = wave >> 6;            // 0..1  (batch halves)
    int n = wave & 63;            // 0..63 (h-column tile)
    int hi = lane >> 4, lm = lane & 15;
    int brow = m * 16 + lm;       // A-fragment row (batch)
    int hcol = n * 16 + lm;       // C-fragment column (h index)

    const __bf16* bptr[4];
    #pragma unroll
    for (int q = 0; q < 4; ++q)
        bptr[q] = whT + (size_t)((q * 64 + n) * 16 + lm) * H_;

    float creg[8], hn[8];
    #pragma unroll
    for (int j = 0; j < 8; ++j) { creg[j] = 0.0f; hn[j] = 0.0f; }

    #pragma unroll 1
    for (int t = 0; t < S_; ++t) {
        const __bf16* hr = hbuf + (t & 1) * (B_ * H_);
        __bf16*       hw = hbuf + ((t + 1) & 1) * (B_ * H_);

        // accumulators seeded from gx fragments (bias already folded in)
        v8f acc[4];
        #pragma unroll
        for (int q = 0; q < 4; ++q) {
            size_t idx = ((((size_t)t * 2 + m) * 256 + (q * 64 + n)) * 32 + lane) * 8;
            v8bf gv = __builtin_nontemporal_load((const v8bf*)(gxf + idx));
            #pragma unroll
            for (int j = 0; j < 8; ++j) acc[q][j] = (float)gv[j];
        }
        if (t + 1 < S_) {  // prefetch next step's gx fragment
            size_t nidx = ((((size_t)(t + 1) * 2 + m) * 256 + n) * 32 + lane) * 8;
            __builtin_prefetch(gxf + nidx, 0, 0);
        }

        const __bf16* arow = hr + (size_t)brow * H_;
        #pragma unroll 2
        for (int kk = 0; kk < H_ / 32; ++kk) {
            int k0 = kk * 32 + hi * 8;
            v8bf a0 = *(const v8bf*)(arow + k0);
            v8bf a1 = *(const v8bf*)(arow + k0 + 16);
            v16bf a;
            #pragma unroll
            for (int j = 0; j < 8; ++j) { a[j] = a0[j]; a[j + 8] = a1[j]; }
            int kb = kk * 32 + hi * 8;
            #pragma unroll
            for (int q = 0; q < 4; ++q) {           // 4 gates share the A fragment
                v8bf b0 = *(const v8bf*)(bptr[q] + kb);
                v8bf b1 = *(const v8bf*)(bptr[q] + kb + 16);
                v16bf b;
                #pragma unroll
                for (int j = 0; j < 8; ++j) { b[j] = b0[j]; b[j + 8] = b1[j]; }
                acc[q] = __builtin_amdgcn_wmma_f32_16x16x32_bf16(false, a, false, b,
                                                                 (short)0, acc[q], false, false);
            }
        }

        // lane-local cell update (reference quirks: sigmoid on candidate, emit o)
        #pragma unroll
        for (int j = 0; j < 8; ++j) {
            int b = m * 16 + j + hi * 8;            // C-fragment row -> batch index
            float f  = fast_sigmoid(acc[0][j]);
            float ig = fast_sigmoid(acc[1][j]);
            float o  = fast_sigmoid(acc[2][j]);
            float cc = fast_sigmoid(acc[3][j]);
            float c  = f * creg[j] + ig * cc;
            creg[j]  = c;
            float h  = o * tanhf(c);
            hn[j]    = h;
            hw[(size_t)b * H_ + hcol] = (__bf16)h;
            __builtin_nontemporal_store(o, &out[(size_t)b * S_ * H_ + (size_t)t * H_ + hcol]);
        }

        // device-wide barrier: monotonic counter, 16 co-resident blocks
        __syncthreads();
        if (threadIdx.x == 0) {
            __threadfence();
            atomicAdd(cnt, 1u);
            unsigned target = (unsigned)(NB * (t + 1));
            volatile unsigned* vc = cnt;
            while (*vc < target) { __builtin_amdgcn_s_sleep(1); }
            __threadfence();
        }
        __syncthreads();
    }

    // tail outputs: ht then ct, each [B, H]
    float* tail = out + (size_t)B_ * S_ * H_;
    #pragma unroll
    for (int j = 0; j < 8; ++j) {
        int b = m * 16 + j + hi * 8;
        tail[(size_t)b * H_ + hcol] = hn[j];
        tail[(size_t)B_ * H_ + (size_t)b * H_ + hcol] = creg[j];
    }
}

extern "C" void kernel_launch(void* const* d_in, const int* in_sizes, int n_in,
                              void* d_out, int out_size, void* d_ws, size_t ws_size,
                              hipStream_t stream) {
    const float* x   = (const float*)d_in[0];
    const float* wxf = (const float*)d_in[1];
    const float* whf = (const float*)d_in[2];
    const float* bhf = (const float*)d_in[3];
    const float* wxi = (const float*)d_in[4];
    const float* whi = (const float*)d_in[5];
    const float* bhi = (const float*)d_in[6];
    const float* wxo = (const float*)d_in[7];
    const float* who = (const float*)d_in[8];
    const float* bho = (const float*)d_in[9];
    const float* wxc = (const float*)d_in[10];
    const float* whc = (const float*)d_in[11];
    const float* bhc = (const float*)d_in[12];
    float* out = (float*)d_out;

    char* ws = (char*)d_ws;
    __bf16*   gxf  = (__bf16*)(ws + OFF_GX);
    __bf16*   wxT  = (__bf16*)(ws + OFF_WXT);
    __bf16*   whT  = (__bf16*)(ws + OFF_WHT);
    __bf16*   xb   = (__bf16*)(ws + OFF_XB);
    float*    bh4  = (float*) (ws + OFF_BH);
    __bf16*   hbuf = (__bf16*)(ws + OFF_H);
    unsigned* cnt  = (unsigned*)(ws + OFF_CNT);

    pack_wx  <<<(4096 * 512       + 255) / 256, 256, 0, stream>>>(wxf, wxi, wxo, wxc, wxT);
    pack_wh  <<<(4096 * 1024      + 255) / 256, 256, 0, stream>>>(whf, whi, who, whc, whT);
    pack_x   <<<(32 * 512 * 512   + 255) / 256, 256, 0, stream>>>(x, xb);
    pack_misc<<<(2 * 32 * 1024    + 255) / 256, 256, 0, stream>>>(bhf, bhi, bho, bhc, bh4, hbuf, cnt);

    // phase 1: 512*2*256 tiles, 1 tile per wave, 8 waves per block
    gx_gemm<<<(512 * 2 * 256) / 8, 256, 0, stream>>>(xb, wxT, bh4, gxf);

    // phase 2: persistent scan, 16 blocks x 8 waves = 128 waves
    lstm_steps<<<NB, 256, 0, stream>>>(gxf, whT, hbuf, out, cnt);
}
